// PitchAttention_37306085933192
// MI455X (gfx1250) — matrix operand order repaired and checked by reading, not compiled, not run on hardware
//
#include <hip/hip_runtime.h>
#include <hip/hip_fp16.h>

typedef __attribute__((ext_vector_type(16))) _Float16 v16h;
typedef __attribute__((ext_vector_type(8)))  float    v8f;

#define S_TOTAL 3072
#define DMODEL  512

#define WMMA_F16(a, b, c) \
  __builtin_amdgcn_wmma_f32_16x16x32_f16(false, (a), false, (b), (short)0, (c), false, false)

__device__ __forceinline__ v8f zero8() {
  v8f z;
#pragma unroll
  for (int i = 0; i < 8; ++i) z[i] = 0.0f;
  return z;
}

// A fragment 16x32 f16 from row-major LDS tile (ISA 7.12.2):
// lane L: row M = L&15; K-halves {hi..hi+7} and {hi+16..hi+23}, hi = (L>>4)*8.
__device__ __forceinline__ v16h frag_a_lds(const _Float16* base, int stride, int lane) {
  const int m  = lane & 15;
  const int hi = (lane >> 4) << 3;
  const _Float16* p = base + m * stride + hi;
  v16h a;
  *reinterpret_cast<uint4*>(&a)       = *reinterpret_cast<const uint4*>(p);
  *(reinterpret_cast<uint4*>(&a) + 1) = *reinterpret_cast<const uint4*>(p + 16);
  return a;
}

// B fragment 32x16 f16 from K-transposed LDS tile Bt[n][k]:
// lane L: col N = L&15; K-halves {hi..hi+7} and {hi+16..hi+23}, hi = (L>>4)*8.
__device__ __forceinline__ v16h frag_b_lds(const _Float16* base, int stride, int lane) {
  const int n  = lane & 15;
  const int hi = (lane >> 4) << 3;
  const _Float16* p = base + n * stride + hi;
  v16h b;
  *reinterpret_cast<uint4*>(&b)       = *reinterpret_cast<const uint4*>(p);
  *(reinterpret_cast<uint4*>(&b) + 1) = *reinterpret_cast<const uint4*>(p + 16);
  return b;
}

// CDNA5 async global->LDS, 16B per lane, tracked by ASYNCcnt (no VGPR round-trip).
// LDS aperture maps addr[31:0] -> LDS byte address (ISA 10.2), so the low 32
// bits of the generic pointer are the VDST operand.
__device__ __forceinline__ void async_b128(const _Float16* lds, const _Float16* g) {
  asm volatile("global_load_async_to_lds_b128 %0, %1, off"
               :: "v"((unsigned)(unsigned long long)lds),
                  "v"((unsigned long long)g)
               : "memory");
}

// ---------------------------------------------------------------------------
// Kernel 1: fused QKV projection (f32 in, f16 out) + pitch bias on Q and K.
// X:[6144,512]  cols 0..511 -> Q, 512..639 -> K, 640..767 -> V (transposed).
// grid (48, 12), block 256 (8 waves; wave w owns 16 rows of the 128-row tile).
// ---------------------------------------------------------------------------
__global__ __launch_bounds__(256) void qkv_kernel(
    const float* __restrict__ X,  const float* __restrict__ Wq,
    const float* __restrict__ Wk, const float* __restrict__ Wv,
    const float* __restrict__ pitch,
    _Float16* __restrict__ qo, _Float16* __restrict__ ko, _Float16* __restrict__ vto) {
  __shared__ __align__(16) _Float16 Xs[128 * 32];
  __shared__ __align__(16) _Float16 Ws[64 * 32];
  const int tid = threadIdx.x, lane = tid & 31, w = tid >> 5;
  const int rowBlock = blockIdx.x * 128;
  const int colBlock = blockIdx.y * 64;

  v8f acc[4];
#pragma unroll
  for (int ct = 0; ct < 4; ++ct) acc[ct] = zero8();

  for (int kt = 0; kt < 16; ++kt) {
    const int kbase = kt * 32;
    // X tile 128x32: f32 -> f16, row-major (A operand)
#pragma unroll
    for (int i = 0; i < 4; ++i) {
      int u = tid + i * 256;          // 0..1023 float4 slots
      int row = u >> 3, c4 = u & 7;
      float4 v = reinterpret_cast<const float4*>(
          X + (size_t)(rowBlock + row) * DMODEL + kbase)[c4];
      _Float16* dst = Xs + row * 32 + c4 * 4;
      dst[0] = (_Float16)v.x; dst[1] = (_Float16)v.y;
      dst[2] = (_Float16)v.z; dst[3] = (_Float16)v.w;
    }
    // W tile, stored K-transposed: Ws[col][k] (B operand)
#pragma unroll
    for (int i = 0; i < 8; ++i) {
      int u = tid + i * 256;          // 0..2047
      int col = u >> 5, kk = u & 31;
      int gc = colBlock + col;
      const float* Wsel; int ldw, c;
      if (gc < 512)      { Wsel = Wq; ldw = 512; c = gc;       }
      else if (gc < 640) { Wsel = Wk; ldw = 128; c = gc - 512; }
      else               { Wsel = Wv; ldw = 128; c = gc - 640; }
      Ws[col * 32 + kk] = (_Float16)Wsel[(size_t)(kbase + kk) * ldw + c];
    }
    __syncthreads();
    v16h a = frag_a_lds(Xs + w * 16 * 32, 32, lane);
#pragma unroll
    for (int ct = 0; ct < 4; ++ct) {
      v16h b = frag_b_lds(Ws + ct * 16 * 32, 32, lane);
      acc[ct] = WMMA_F16(a, b, acc[ct]);
    }
    __syncthreads();
  }

  const int hi8 = (lane >> 4) << 3, cn = lane & 15;
#pragma unroll
  for (int ct = 0; ct < 4; ++ct) {
    int gc = colBlock + ct * 16 + cn;
#pragma unroll
    for (int r = 0; r < 8; ++r) {
      int row = w * 16 + hi8 + r;
      int n = rowBlock + row;
      int b = n / S_TOTAL, s = n % S_TOTAL, p = n % 12;
      float val = acc[ct][r];
      if (gc < 512) {                       // Q + bias, layout [B,H,S,64]
        int h = gc >> 6, dc = gc & 63;
        val += pitch[p * 64 + dc];
        qo[((size_t)(b * 8 + h) * S_TOTAL + s) * 64 + dc] = (_Float16)val;
      } else if (gc < 640) {                // K + bias, layout [B,Hkv,S,64]
        int kvh = (gc - 512) >> 6, dc = (gc - 512) & 63;
        val += pitch[p * 64 + dc];
        ko[((size_t)(b * 2 + kvh) * S_TOTAL + s) * 64 + dc] = (_Float16)val;
      } else {                              // V transposed, layout [B,Hkv,64,S]
        int kvh = (gc - 640) >> 6, dc = (gc - 640) & 63;
        vto[((size_t)(b * 2 + kvh) * 64 + dc) * S_TOTAL + s] = (_Float16)val;
      }
    }
  }
}

// ---------------------------------------------------------------------------
// Kernel 2: flash attention. grid (24 q-tiles, 16 (b,h)), block 256.
// Wave w owns 16 query rows; loop over 48 key chunks of 64 w/ online softmax.
// K/V tiles double-buffered via async global->LDS (ASYNCcnt), Q frags hoisted.
// ---------------------------------------------------------------------------
__global__ __launch_bounds__(256) void attn_kernel(
    const _Float16* __restrict__ Q, const _Float16* __restrict__ K,
    const _Float16* __restrict__ Vt, _Float16* __restrict__ Aout) {
  __shared__ __align__(16) _Float16 Qs[128 * 64];      // 16 KB
  __shared__ __align__(16) _Float16 Ks[2][64 * 64];    // 16 KB  [key][d]
  __shared__ __align__(16) _Float16 Vs[2][64 * 64];    // 16 KB  [d][key]
  __shared__ __align__(16) _Float16 Ps[8][16 * 64];    // 16 KB  per-wave P tile
  const int tid = threadIdx.x, lane = tid & 31, w = tid >> 5;
  const int hi8 = (lane >> 4) << 3, cn = lane & 15;
  const int bh = blockIdx.y, b = bh >> 3, h = bh & 7, kvh = h >> 2;
  const int qtile = blockIdx.x;
  const size_t qbase  = ((size_t)bh * S_TOTAL + qtile * 128) * 64;
  const _Float16* Kb  = K  + (size_t)(b * 2 + kvh) * S_TOTAL * 64;
  const _Float16* Vb  = Vt + (size_t)(b * 2 + kvh) * 64 * S_TOTAL;

  // issue async stage of chunk 0 into buffer 0 (overlaps the Q copy below)
  {
    const _Float16* ks = Kb;          // chunk 0
    async_b128(&Ks[0][tid * 8],         ks + tid * 8);
    async_b128(&Ks[0][(tid + 256) * 8], ks + (tid + 256) * 8);
#pragma unroll
    for (int i = 0; i < 2; ++i) {
      int u = tid + i * 256;          // 0..511
      int d = u >> 3, c8 = u & 7;
      async_b128(&Vs[0][d * 64 + c8 * 8], Vb + (size_t)d * S_TOTAL + c8 * 8);
    }
  }
  // Q tile: 8192 halves = 1024 uint4 (synchronous; consumed immediately)
  {
    const uint4* src = reinterpret_cast<const uint4*>(Q + qbase);
    uint4* dst = reinterpret_cast<uint4*>(Qs);
#pragma unroll
    for (int i = 0; i < 4; ++i) dst[tid + i * 256] = src[tid + i * 256];
  }
  v8f o_acc[4];
#pragma unroll
  for (int ct = 0; ct < 4; ++ct) o_acc[ct] = zero8();
  float mrow[8], lrow[8];
#pragma unroll
  for (int r = 0; r < 8; ++r) { mrow[r] = -1e30f; lrow[r] = 0.0f; }
  __syncthreads();                    // Q visible; chunk-0 async still in flight

  // hoisted loop-invariant Q fragments (16x32 each, d = 0..31 and 32..63)
  const v16h qa0 = frag_a_lds(Qs + w * 16 * 64,      64, lane);
  const v16h qa1 = frag_a_lds(Qs + w * 16 * 64 + 32, 64, lane);

  for (int ck = 0; ck < 48; ++ck) {
    const int buf = ck & 1;
    const _Float16* KsB = Ks[buf];
    const _Float16* VsB = Vs[buf];
    // prefetch chunk ck+1 into the spare buffer (freed by last iter's barrier)
    if (ck + 1 < 48) {
      const _Float16* ks = Kb + (size_t)(ck + 1) * 64 * 64;
      _Float16* kd = Ks[buf ^ 1];
      _Float16* vd = Vs[buf ^ 1];
      async_b128(&kd[tid * 8],         ks + tid * 8);
      async_b128(&kd[(tid + 256) * 8], ks + (tid + 256) * 8);
#pragma unroll
      for (int i = 0; i < 2; ++i) {
        int u = tid + i * 256;
        int d = u >> 3, c8 = u & 7;
        async_b128(&vd[d * 64 + c8 * 8],
                   Vb + (size_t)d * S_TOTAL + (ck + 1) * 64 + c8 * 8);
      }
      // async loads complete in order: <=4 outstanding => current buf is done
      asm volatile("s_wait_asynccnt 4" ::: "memory");
    } else {
      asm volatile("s_wait_asynccnt 0" ::: "memory");
    }
    __syncthreads();

    // scores S = Q * K^T * 0.125
    v8f st_acc[4];
#pragma unroll
    for (int st = 0; st < 4; ++st) {
      v8f sa = zero8();
      sa = WMMA_F16(qa0, frag_b_lds(KsB + st * 16 * 64,      64, lane), sa);
      sa = WMMA_F16(qa1, frag_b_lds(KsB + st * 16 * 64 + 32, 64, lane), sa);
#pragma unroll
      for (int r = 0; r < 8; ++r) sa[r] *= 0.125f;
      st_acc[st] = sa;
    }
    // row max over 64 keys: 4 tiles then 16-lane half-wave reduction
    float mnew[8];
#pragma unroll
    for (int r = 0; r < 8; ++r) {
      float v = fmaxf(fmaxf(st_acc[0][r], st_acc[1][r]),
                      fmaxf(st_acc[2][r], st_acc[3][r]));
#pragma unroll
      for (int off = 8; off >= 1; off >>= 1) v = fmaxf(v, __shfl_xor(v, off, 32));
      mnew[r] = fmaxf(mrow[r], v);
    }
#pragma unroll
    for (int r = 0; r < 8; ++r) {         // online rescale
      float corr = __expf(mrow[r] - mnew[r]);
      mrow[r] = mnew[r];
      lrow[r] *= corr;
#pragma unroll
      for (int ct = 0; ct < 4; ++ct) o_acc[ct][r] *= corr;
    }
    float rs[8];
#pragma unroll
    for (int r = 0; r < 8; ++r) rs[r] = 0.0f;
#pragma unroll
    for (int st = 0; st < 4; ++st)
#pragma unroll
      for (int r = 0; r < 8; ++r) {
        float pv = __expf(st_acc[st][r] - mrow[r]);
        rs[r] += pv;
        Ps[w][(hi8 + r) * 64 + st * 16 + cn] = (_Float16)pv;  // C-layout -> LDS
      }
#pragma unroll
    for (int r = 0; r < 8; ++r) {
      float v = rs[r];
#pragma unroll
      for (int off = 8; off >= 1; off >>= 1) v += __shfl_xor(v, off, 32);
      lrow[r] += v;
    }
    asm volatile("s_wait_dscnt 0" ::: "memory");  // P visible for A-frag reads

    // O += P * V
#pragma unroll
    for (int kk = 0; kk < 2; ++kk) {
      v16h a = frag_a_lds(&Ps[w][kk * 32], 64, lane);
#pragma unroll
      for (int ct = 0; ct < 4; ++ct) {
        v16h bfr = frag_b_lds(VsB + ct * 16 * 64 + kk * 32, 64, lane);
        o_acc[ct] = WMMA_F16(a, bfr, o_acc[ct]);
      }
    }
    __syncthreads();   // all waves done reading buf -> spare buffer reusable
  }

#pragma unroll
  for (int ct = 0; ct < 4; ++ct)
#pragma unroll
    for (int r = 0; r < 8; ++r) {
      float val = o_acc[ct][r] / lrow[r];
      int sq = qtile * 128 + w * 16 + hi8 + r;
      int n = b * S_TOTAL + sq;
      int col = h * 64 + ct * 16 + cn;
      Aout[(size_t)n * 512 + col] = (_Float16)val;   // [B,T,P, H*64]
    }
}

// ---------------------------------------------------------------------------
// Kernel 3: output projection attn(f16)[6144,512] x Wo(f32->f16) -> f32 out.
// grid (48, 8), block 256.
// ---------------------------------------------------------------------------
__global__ __launch_bounds__(256) void proj_kernel(
    const _Float16* __restrict__ A, const float* __restrict__ Wo,
    float* __restrict__ out) {
  __shared__ __align__(16) _Float16 Xs[128 * 32];
  __shared__ __align__(16) _Float16 Ws[64 * 32];
  const int tid = threadIdx.x, lane = tid & 31, w = tid >> 5;
  const int rowBlock = blockIdx.x * 128, colBlock = blockIdx.y * 64;
  v8f acc[4];
#pragma unroll
  for (int ct = 0; ct < 4; ++ct) acc[ct] = zero8();

  for (int kt = 0; kt < 16; ++kt) {
    const int kbase = kt * 32;
#pragma unroll
    for (int i = 0; i < 2; ++i) {
      int u = tid + i * 256;          // 0..511 uint4 slots
      int row = u >> 2, c4 = u & 3;
      reinterpret_cast<uint4*>(Xs)[row * 4 + c4] =
          reinterpret_cast<const uint4*>(A + (size_t)(rowBlock + row) * 512 + kbase)[c4];
    }
#pragma unroll
    for (int i = 0; i < 8; ++i) {
      int u = tid + i * 256;
      int col = u >> 5, kk = u & 31;
      Ws[col * 32 + kk] = (_Float16)Wo[(size_t)(kbase + kk) * 512 + colBlock + col];
    }
    __syncthreads();
    v16h a = frag_a_lds(Xs + w * 16 * 32, 32, lane);
#pragma unroll
    for (int ct = 0; ct < 4; ++ct) {
      v16h bfr = frag_b_lds(Ws + ct * 16 * 32, 32, lane);
      acc[ct] = WMMA_F16(a, bfr, acc[ct]);
    }
    __syncthreads();
  }
  const int hi8 = (lane >> 4) << 3, cn = lane & 15;
#pragma unroll
  for (int ct = 0; ct < 4; ++ct)
#pragma unroll
    for (int r = 0; r < 8; ++r)
      out[(size_t)(rowBlock + w * 16 + hi8 + r) * 512 + colBlock + ct * 16 + cn] =
          acc[ct][r];
}

extern "C" void kernel_launch(void* const* d_in, const int* in_sizes, int n_in,
                              void* d_out, int out_size, void* d_ws, size_t ws_size,
                              hipStream_t stream) {
  const float* hs = (const float*)d_in[0];
  const float* Wq = (const float*)d_in[1];
  const float* Wk = (const float*)d_in[2];
  const float* Wv = (const float*)d_in[3];
  const float* Wo = (const float*)d_in[4];
  const float* pt = (const float*)d_in[5];

  _Float16* qws  = (_Float16*)d_ws;                          // [2,8,3072,64]
  _Float16* kws  = qws  + (size_t)2 * 8 * S_TOTAL * 64;      // [2,2,3072,64]
  _Float16* vtws = kws  + (size_t)2 * 2 * S_TOTAL * 64;      // [2,2,64,3072]
  _Float16* aws  = vtws + (size_t)2 * 2 * S_TOTAL * 64;      // [6144,512]

  qkv_kernel <<<dim3(48, 12), 256, 0, stream>>>(hs, Wq, Wk, Wv, pt, qws, kws, vtws);
  attn_kernel<<<dim3(24, 16), 256, 0, stream>>>(qws, kws, vtws, aws);
  proj_kernel<<<dim3(48, 8),  256, 0, stream>>>(aws, Wo, (float*)d_out);
}
